// WordSpeechBinaryFusion_4896262718143
// MI455X (gfx1250) — matrix operand order repaired and checked by reading, not compiled, not run on hardware
//
#include <hip/hip_runtime.h>

// -------------------------------------------------------------------------
// WordSpeechBinaryFusion on MI455X (gfx1250, wave32)
//
// Persistent kernel: one 256-thread workgroup (8 wave32s) per batch row runs
// the entire data-dependent while-loop on device (graph-capture safe).
//   - score GEMV (u = x.w_lo, v = x.w_hi) : LDS-cached weights, wave reduce
//   - greedy non-overlap scan + compaction map : sequential on lane 0 (LDS)
//   - combine GEMM only for fused pairs, 16-pair tiles via
//     v_wmma_f32_16x16x32_bf16 (bf16 inputs, fp32 accumulate)
//   - ping-pong between d_out row and d_ws row; final pass zero-pads tail
//
// d_ws layout:  [0, B*S*D floats)              pong buffer (32 MB)
//               [B*S*D floats, +512*256 bf16)  W_combine pre-swizzled into
//                                              WMMA B-fragment order (256 KB)
// -------------------------------------------------------------------------

typedef __attribute__((ext_vector_type(16))) __bf16 v16bf;
typedef __attribute__((ext_vector_type(8)))  float  v8f;

#define BATCH 16
#define SEQ   2048
#define DIM   256
#define TWOD  512
#define THR   0.5f

__device__ __forceinline__ __bf16 f2bf(float f) {
    unsigned u = __builtin_bit_cast(unsigned, f);
    unsigned r = u + 0x7FFFu + ((u >> 16) & 1u);   // round-to-nearest-even
    unsigned short h = (unsigned short)(r >> 16);
    return __builtin_bit_cast(__bf16, h);
}

__global__ __launch_bounds__(256)
void fuse_persistent(const float* __restrict__ xin,     // [B,S,D]
                     const float* __restrict__ Wscore,  // [2D,1]
                     const float* __restrict__ bscore,  // [1]
                     const float* __restrict__ Wcomb,   // [2D,D]
                     const float* __restrict__ bcomb,   // [D]
                     float* __restrict__ out,           // [B,S,D]
                     float* __restrict__ ws)
{
    __shared__ float s_wlo[DIM], s_whi[DIM];
    __shared__ float s_u[SEQ], s_v[SEQ];
    __shared__ unsigned char s_keep[SEQ];
    __shared__ int s_map[SEQ];                 // dst -> src (bit30 = fused-pair left idx)
    __shared__ int s_pairL[SEQ / 2], s_pairD[SEQ / 2];
    __shared__ __bf16 s_abf[16 * TWOD];        // 16-pair A tile, bf16, row-major
    __shared__ int s_newLen, s_np, s_any;

    const int b    = blockIdx.x;
    const int tid  = threadIdx.x;
    const int lane = tid & 31;
    const int wv   = tid >> 5;

    float*  row0 = out + (size_t)b * SEQ * DIM;                   // ping
    float*  row1 = ws  + (size_t)b * SEQ * DIM;                   // pong
    __bf16* Wsw  = (__bf16*)(ws + (size_t)BATCH * SEQ * DIM);     // swizzled W (shared)

    // ---- phase 0: copy input row into ping buffer ----
    {
        const float4* src = (const float4*)(xin + (size_t)b * SEQ * DIM);
        float4*       dst = (float4*)row0;
        for (int i = tid; i < SEQ * DIM / 4; i += 256) dst[i] = src[i];
    }
    // ---- phase 0b: W_combine -> bf16, pre-swizzled into WMMA B-fragment order.
    // All 16 blocks write identical bytes (benign race); each block reads only
    // after writing every entry itself. Layout: [nt][kt][lane][e] with
    //   K = kt*32 + (lane>>4)*16 + e,  N = nt*16 + (lane&15)
    for (int idx = tid; idx < 16 * 16 * 32 * 16; idx += 256) {
        int e  = idx & 15;
        int ln = (idx >> 4) & 31;
        int kt = (idx >> 9) & 15;
        int nt = idx >> 13;
        int k  = kt * 32 + (ln >> 4) * 16 + e;
        int n  = nt * 16 + (ln & 15);
        Wsw[idx] = f2bf(Wcomb[k * DIM + n]);
    }
    for (int j = tid; j < DIM; j += 256) {
        s_wlo[j] = Wscore[j];
        s_whi[j] = Wscore[DIM + j];
    }
    const float bsc = bscore[0];
    __syncthreads();

    float* cur = row0;
    float* nxt = row1;
    int len = SEQ;

    while (true) {
        // ---- dual dot products: u[s] = x[s].w_lo, v[s] = x[s].w_hi ----
        for (int s = wv; s < len; s += 8) {
            const float* r = cur + (size_t)s * DIM;
            float su = 0.f, sv = 0.f;
            for (int j = lane; j < DIM; j += 32) {
                float xv = r[j];
                su += xv * s_wlo[j];
                sv += xv * s_whi[j];
            }
            for (int off = 16; off > 0; off >>= 1) {
                su += __shfl_down(su, off, 32);
                sv += __shfl_down(sv, off, 32);
            }
            if (lane == 0) { s_u[s] = su; s_v[s] = sv; }
        }
        __syncthreads();

        // ---- pair selection ----
        for (int i = tid; i < len - 1; i += 256)
            s_keep[i] = (s_u[i] + s_v[i + 1] + bsc > THR) ? 1 : 0;
        __syncthreads();

        // ---- greedy non-overlap + compaction map (sequential recurrence) ----
        if (tid == 0) {
            int any = 0;
            unsigned char pk = 0;
            for (int i = 0; i < len - 1; i++) {
                unsigned char k = (unsigned char)(s_keep[i] & (pk ^ 1));
                s_keep[i] = k; pk = k; any |= k;
            }
            int dst = 0, np = 0, s = 0;
            while (s < len) {
                if (s < len - 1 && s_keep[s]) {
                    s_map[dst] = 0x40000000 | s;
                    s_pairL[np] = s; s_pairD[np] = dst;
                    np++; dst++; s += 2;
                } else {
                    s_map[dst] = s;
                    dst++; s += 1;
                }
            }
            s_newLen = dst; s_np = np; s_any = any;
        }
        __syncthreads();
        if (s_any == 0) break;
        const int newLen = s_newLen;
        const int np     = s_np;

        // ---- copy un-fused rows cur -> nxt (float4) ----
        for (int idx = tid; idx < newLen * (DIM / 4); idx += 256) {
            int d = idx / (DIM / 4), c = idx % (DIM / 4);
            int m = s_map[d];
            if (!(m & 0x40000000))
                ((float4*)(nxt + (size_t)d * DIM))[c] =
                    ((const float4*)(cur + (size_t)m * DIM))[c];
        }

        // ---- fused rows: 16-pair tiles, D = [XL|XR] @ W_combine via WMMA ----
        for (int t0 = 0; t0 < np; t0 += 16) {
            int cnt = (np - t0 < 16) ? (np - t0) : 16;
            __syncthreads();   // protect s_abf reuse across tiles
            // stage A tile [16 x 512] bf16 (zero-padded rows)
            for (int idx = tid; idx < 16 * TWOD; idx += 256) {
                int r = idx >> 9, c = idx & (TWOD - 1);
                float v = 0.f;
                if (r < cnt) {
                    int sL = s_pairL[t0 + r];
                    v = (c < DIM) ? cur[(size_t)sL * DIM + c]
                                  : cur[(size_t)(sL + 1) * DIM + (c - DIM)];
                }
                s_abf[idx] = f2bf(v);
            }
            __syncthreads();

            const int half = lane >> 4;
            const int mrow = lane & 15;
            // each wave owns 2 of the 16 N-tiles; uniform control => EXEC all-1s
            for (int nt = wv; nt < 16; nt += 8) {
                v8f acc = {};
                for (int kt = 0; kt < 16; kt++) {
                    v16bf a, bb;
                    // A fragment: lane -> M = lane&15, K = kt*32 + half*16 + e
                    const __bf16* ap = &s_abf[mrow * TWOD + kt * 32 + half * 16];
#pragma unroll
                    for (int e = 0; e < 8; e++) { a[e] = ap[e]; a[e + 8] = ap[8 + e]; }
                    // B fragment: pre-swizzled, one aligned 32B load per lane
                    bb = *(const v16bf*)&Wsw[(((nt * 16 + kt) * 32) + lane) * 16];
                    acc = __builtin_amdgcn_wmma_f32_16x16x32_bf16(
                        /*neg_a=*/false, a, /*neg_b=*/false, bb,
                        /*c_mod=*/(short)0, acc, /*reuse_a=*/false, /*reuse_b=*/false);
                }
                int col  = nt * 16 + (lane & 15);
                float bias = bcomb[col];
#pragma unroll
                for (int r = 0; r < 8; r++) {
                    int M = r + 8 * half;   // D row M = r + 8*(lane>=16)
                    if (M < cnt)
                        nxt[(size_t)s_pairD[t0 + M] * DIM + col] = acc[r] + bias;
                }
            }
        }
        __syncthreads();

        // swap ping-pong
        float* tmp = cur; cur = nxt; nxt = tmp;
        len = newLen;
    }

    // ---- finalize: compacted rows + zero tail into out row ----
    __syncthreads();
    float* orow = out + (size_t)b * SEQ * DIM;
    const float4 z4 = make_float4(0.f, 0.f, 0.f, 0.f);
    for (int idx = tid; idx < SEQ * DIM / 4; idx += 256) {
        int s = idx / (DIM / 4);
        float4 v = (s < len) ? ((const float4*)cur)[idx] : z4;
        ((float4*)orow)[idx] = v;
    }
}

extern "C" void kernel_launch(void* const* d_in, const int* in_sizes, int n_in,
                              void* d_out, int out_size, void* d_ws, size_t ws_size,
                              hipStream_t stream) {
    (void)in_sizes; (void)n_in; (void)out_size;
    const float* frame_input = (const float*)d_in[0];   // [16,2048,256] f32
    const float* W_score     = (const float*)d_in[1];   // [512,1]       f32
    const float* b_score     = (const float*)d_in[2];   // [1]           f32
    const float* W_combine   = (const float*)d_in[3];   // [512,256]     f32
    const float* b_combine   = (const float*)d_in[4];   // [256]         f32
    float*       out         = (float*)d_out;           // [16,2048,256] f32

    // needs ws_size >= B*S*D*4 + 512*256*2 bytes (~33.8 MB)
    (void)ws_size;
    fuse_persistent<<<dim3(BATCH), dim3(256), 0, stream>>>(
        frame_input, W_score, b_score, W_combine, b_combine, out, (float*)d_ws);
}
